// MSAPairWeightedAveraging_42786464202818
// MI455X (gfx1250) — matrix operand order repaired and checked by reading, not compiled, not run on hardware
//
#include <hip/hip_runtime.h>
#include <hip/hip_bf16.h>
#include <stdint.h>

// ---------------------------------------------------------------- constants
#define S_ 256
#define N_ 384
#define CM_ 64
#define CZ_ 128
#define C_ 32
#define H_ 8
#define HC_ 256   // H*C

typedef __bf16 bf16_t;
typedef __attribute__((ext_vector_type(16))) __bf16 v16bf;
typedef __attribute__((ext_vector_type(8)))  __bf16 v8bf;
typedef __attribute__((ext_vector_type(8)))  float  v8f;

// async-to-LDS builtin operand types (signature revealed by clang diagnostics:
// param0 = AS1 pointer to int __vector_size__(16), param1 = AS3 counterpart)
typedef int v4i __attribute__((vector_size(16)));
typedef __attribute__((address_space(1))) v4i v4i_as1;
typedef __attribute__((address_space(3))) v4i v4i_as3;

// feature probes (evaluated independently per host/device pass)
#if defined(__has_builtin)
#  if __has_builtin(__builtin_amdgcn_global_load_async_to_lds_b128)
#    define HAVE_ASYNC_LDS 1
#  endif
#  if __has_builtin(__builtin_amdgcn_s_wait_asynccnt)
#    define HAVE_WAIT_ASYNC 1
#  endif
#endif

#if defined(HAVE_ASYNC_LDS)
template <int N>
__device__ __forceinline__ void wait_asynccnt_le() {
#if defined(HAVE_WAIT_ASYNC)
    __builtin_amdgcn_s_wait_asynccnt(N);       // constant integer required
#else
    asm volatile("s_wait_asynccnt %0" :: "i"(N) : "memory");
#endif
}
#endif

// ------------------------------------------------------------- conversions
__device__ __forceinline__ bf16_t f2bf(float x) {
    uint32_t u = __builtin_bit_cast(uint32_t, x);
    uint32_t r = u + 0x7FFFu + ((u >> 16) & 1u);       // round-to-nearest-even
    unsigned short h = (unsigned short)(r >> 16);
    return __builtin_bit_cast(bf16_t, h);
}
__device__ __forceinline__ float bf2f(bf16_t h) {
    uint32_t u = ((uint32_t)__builtin_bit_cast(unsigned short, h)) << 16;
    return __builtin_bit_cast(float, u);
}

// ---------------------------------------------------------- WMMA fragments
// A fragment (16x32 bf16), row-major source, leading dim lda (in halves).
// Contiguous K order (same as load_b_frag): lanes 0-15 -> K k0+0..15, lanes 16-31 -> K k0+16..31.
__device__ __forceinline__ v16bf load_a_frag(const bf16_t* base, int lda,
                                             int row0, int k0, int lane) {
    int r  = row0 + (lane & 15);
    int kb = k0 + ((lane & 16) ? 16 : 0);
    const bf16_t* p = base + (size_t)r * lda + kb;
    v8bf lo = *(const v8bf*)(p);
    v8bf hi = *(const v8bf*)(p + 8);
    v16bf a;
#pragma unroll
    for (int e = 0; e < 8; ++e) { a[e] = lo[e]; a[8 + e] = hi[e]; }
    return a;
}

// B fragment (32x16 bf16): B[k,n] = colbase[n*ldk + k]  (K contiguous per column).
// Contiguous K order (same as load_a_frag): column = lane&15, lanes 0-15 hold K k0+0..15, lanes 16-31 K k0+16..31.
__device__ __forceinline__ v16bf load_b_frag(const bf16_t* colbase, int ldk,
                                             int n0, int k0, int lane) {
    int n  = n0 + (lane & 15);
    int kb = k0 + ((lane & 16) ? 16 : 0);
    return *(const v16bf*)(colbase + (size_t)n * ldk + kb);
}

__device__ __forceinline__ v8f wmma_bf16(v16bf a, v16bf b, v8f c) {
    return __builtin_amdgcn_wmma_f32_16x16x32_bf16(
        /*neg_a=*/false, a, /*neg_b=*/false, b,
        /*c_mod=*/(short)0, c, /*reuse_a=*/false, /*reuse_b=*/false);
}

// ================================================================ stage A
// b[i,j,h] = LN(z[i,j,:]) @ w_b  (+ mask); softmax over j;
// store bf16 Wb[h][i][j].  One block per i; wave w handles rows j = w, w+8, ...
// z rows are streamed through LDS with GLOBAL_LOAD_ASYNC_TO_LDS_B128
// (double-buffered per wave) so the next row's fetch overlaps this row's math.
__device__ __forceinline__ void stage_a_row(float4 zv, int j, int lane,
                                            const float* lnw_s, const float* lnb_s,
                                            const float* wb_s, const int* mask,
                                            float (*bmat)[H_])
{
    float s  = zv.x + zv.y + zv.z + zv.w;
    float sq = zv.x * zv.x + zv.y * zv.y + zv.z * zv.z + zv.w * zv.w;
#pragma unroll
    for (int t = 16; t >= 1; t >>= 1) { s += __shfl_xor(s, t, 32); sq += __shfl_xor(sq, t, 32); }
    const float mu  = s * (1.0f / CZ_);
    const float inv = rsqrtf(sq * (1.0f / CZ_) - mu * mu + 1e-5f);
    float zn[4];
#pragma unroll
    for (int k = 0; k < 4; ++k) {
        const int c = lane * 4 + k;
        float raw = (k == 0) ? zv.x : (k == 1) ? zv.y : (k == 2) ? zv.z : zv.w;
        zn[k] = (raw - mu) * inv * lnw_s[c] + lnb_s[c];
    }
    const float bm = mask[j] ? 0.0f : -1e9f;
#pragma unroll
    for (int h = 0; h < H_; ++h) {
        float a = 0.0f;
#pragma unroll
        for (int k = 0; k < 4; ++k) a += zn[k] * wb_s[(lane * 4 + k) * H_ + h];
#pragma unroll
        for (int t = 16; t >= 1; t >>= 1) a += __shfl_xor(a, t, 32);
        if (lane == h) bmat[j][h] = a + bm;
    }
}

__global__ void __launch_bounds__(256)
stage_a(const float* __restrict__ z, const int* __restrict__ mask,
        const float* __restrict__ lnw, const float* __restrict__ lnb,
        const float* __restrict__ w_b, bf16_t* __restrict__ Wb)
{
    const int i    = blockIdx.x;
    const int tid  = threadIdx.x;
    const int wave = tid >> 5, lane = tid & 31;

    __shared__ float wb_s[CZ_ * H_];
    __shared__ float lnw_s[CZ_], lnb_s[CZ_];
    __shared__ float bmat[N_][H_];
#if defined(HAVE_ASYNC_LDS)
    __shared__ __align__(16) float zbuf[8][2][CZ_];    // per-wave double buffer
#endif

    for (int idx = tid; idx < CZ_ * H_; idx += 256) wb_s[idx] = w_b[idx];
    for (int idx = tid; idx < CZ_; idx += 256) { lnw_s[idx] = lnw[idx]; lnb_s[idx] = lnb[idx]; }
    __syncthreads();

#if defined(HAVE_ASYNC_LDS)
    {
        // async row streamer: lane l copies 16B -> zbuf[wave][buf][l*4..l*4+3]
        auto issue = [&](int j, int buf) {
            const float* gsrc = z + ((size_t)i * N_ + j) * CZ_ + lane * 4;
            float* ldst = &zbuf[wave][buf][lane * 4];
            v4i_as1* gp = (v4i_as1*)(uintptr_t)(const void*)gsrc;
            v4i_as3* lp = (v4i_as3*)(uint32_t)(uintptr_t)(void*)ldst;
            __builtin_amdgcn_global_load_async_to_lds_b128(gp, lp, 0, 0);
        };
        int buf = 0;
        issue(wave, buf);
        for (int j = wave; j < N_; j += 8) {
            const bool more = (j + 8) < N_;
            if (more) { issue(j + 8, buf ^ 1); wait_asynccnt_le<1>(); }
            else      { wait_asynccnt_le<0>(); }
            const float4 zv = *(const float4*)&zbuf[wave][buf][lane * 4];
            stage_a_row(zv, j, lane, lnw_s, lnb_s, wb_s, mask, bmat);
            buf ^= 1;
        }
    }
#else
    for (int j = wave; j < N_; j += 8) {
        const float4 zv = *(const float4*)(z + ((size_t)i * N_ + j) * CZ_ + lane * 4);
        stage_a_row(zv, j, lane, lnw_s, lnb_s, wb_s, mask, bmat);
    }
#endif
    __syncthreads();

    // softmax over j for h = wave (12 j's per lane)
    {
        const int h = wave;
        float mx = -1e30f;
        for (int j = lane; j < N_; j += 32) mx = fmaxf(mx, bmat[j][h]);
#pragma unroll
        for (int t = 16; t >= 1; t >>= 1) mx = fmaxf(mx, __shfl_xor(mx, t, 32));
        float sum = 0.0f;
        for (int j = lane; j < N_; j += 32) sum += __expf(bmat[j][h] - mx);
#pragma unroll
        for (int t = 16; t >= 1; t >>= 1) sum += __shfl_xor(sum, t, 32);
        const float rs = 1.0f / sum;
        bf16_t* wp = Wb + ((size_t)h * N_ + i) * N_;
        for (int j = lane; j < N_; j += 32) wp[j] = f2bf(__expf(bmat[j][h] - mx) * rs);
    }
}

// ================================================================ stage B1
// m_ln = LayerNorm(m) over C_M=64, stored bf16 row-major [(s*N+j)][64].
__global__ void __launch_bounds__(256)
stage_b1(const float* __restrict__ m, const float* __restrict__ lnw,
         const float* __restrict__ lnb, uint32_t* __restrict__ mln_pk)
{
    const int row  = blockIdx.x * 8 + (threadIdx.x >> 5);
    const int lane = threadIdx.x & 31;
    const float2 v = *(const float2*)(m + (size_t)row * CM_ + lane * 2);
    float s  = v.x + v.y;
    float sq = v.x * v.x + v.y * v.y;
#pragma unroll
    for (int t = 16; t >= 1; t >>= 1) { s += __shfl_xor(s, t, 32); sq += __shfl_xor(sq, t, 32); }
    const float mu  = s * (1.0f / CM_);
    const float inv = rsqrtf(sq * (1.0f / CM_) - mu * mu + 1e-5f);
    const float2 wv = *(const float2*)(lnw + lane * 2);
    const float2 bv = *(const float2*)(lnb + lane * 2);
    const bf16_t lo = f2bf((v.x - mu) * inv * wv.x + bv.x);
    const bf16_t hi = f2bf((v.y - mu) * inv * wv.y + bv.y);
    const uint32_t pk = (uint32_t)__builtin_bit_cast(unsigned short, lo) |
                        ((uint32_t)__builtin_bit_cast(unsigned short, hi) << 16);
    mln_pk[(size_t)row * (CM_ / 2) + lane] = pk;
}

// ================================================================ stage B2
// v = m_ln @ w_v  -> scatter-store transposed vT[h][s][c][j] (bf16)
// g = sigmoid(m_ln @ w_g) -> gbuf[(s*N+j)][hc] (bf16)
__global__ void __launch_bounds__(256)
stage_b2(const bf16_t* __restrict__ mln, const float* __restrict__ w_v,
         const float* __restrict__ w_g, bf16_t* __restrict__ vT,
         bf16_t* __restrict__ gbuf)
{
    __shared__ __align__(32) bf16_t wvT[HC_ * CM_];   // [hc][k] 32KB
    __shared__ __align__(32) bf16_t wgT[HC_ * CM_];   // [hc][k] 32KB
    const int tid = threadIdx.x;
    for (int idx = tid; idx < HC_ * CM_; idx += 256) {
        const int hc = idx >> 6, k = idx & 63;
        wvT[idx] = f2bf(w_v[k * HC_ + hc]);
        wgT[idx] = f2bf(w_g[k * HC_ + hc]);
    }
    __syncthreads();

    const int wave = tid >> 5, lane = tid & 31;
    const int r0   = blockIdx.x * 128 + wave * 16;      // row = s*N + j
    const v16bf A0 = load_a_frag(mln, CM_, r0, 0, lane);
    const v16bf A1 = load_a_frag(mln, CM_, r0, 32, lane);
    const int rowM = (lane & 16) ? 8 : 0;
    const int ncol = lane & 15;

    for (int n = 0; n < 16; ++n) {                      // v projection
        v8f acc = {};
        acc = wmma_bf16(A0, load_b_frag(wvT, CM_, n * 16, 0, lane), acc);
        acc = wmma_bf16(A1, load_b_frag(wvT, CM_, n * 16, 32, lane), acc);
        const int hc = n * 16 + ncol, h = hc >> 5, c = hc & 31;
#pragma unroll
        for (int e = 0; e < 8; ++e) {
            const int row = r0 + rowM + e;
            const int ss = row / N_, j = row - ss * N_;
            vT[(((size_t)h * S_ + ss) * C_ + c) * N_ + j] = f2bf(acc[e]);
        }
    }
    for (int n = 0; n < 16; ++n) {                      // gate projection
        v8f acc = {};
        acc = wmma_bf16(A0, load_b_frag(wgT, CM_, n * 16, 0, lane), acc);
        acc = wmma_bf16(A1, load_b_frag(wgT, CM_, n * 16, 32, lane), acc);
        const int hc = n * 16 + ncol;
#pragma unroll
        for (int e = 0; e < 8; ++e) {
            const int row = r0 + rowM + e;
            gbuf[(size_t)row * HC_ + hc] = f2bf(1.0f / (1.0f + __expf(-acc[e])));
        }
    }
}

// ================================================================ stage C
// o[s,i,h,c] = sum_j Wb[h][i][j] * vT[h][s][c][j]; gated in-place: g *= o.
// Block: 16 s x 16 i, wave = h. A-frags (12 K-steps of Wb) hoisted in
// registers. Both c-halves accumulated in parallel (2 independent WMMA
// chains per K step) for ILP; next-s vT tile prefetched while computing.
__global__ void __launch_bounds__(256)
stage_c(const bf16_t* __restrict__ Wb, const bf16_t* __restrict__ vT,
        bf16_t* __restrict__ gbuf)
{
    const int tid = threadIdx.x;
    const int h = tid >> 5, lane = tid & 31;
    const int s0 = blockIdx.x * 16;
    const int i0 = blockIdx.y * 16;

    const bf16_t* wbase = Wb + (size_t)h * N_ * N_;     // [i][j]
    v16bf A[12];
#pragma unroll
    for (int k = 0; k < 12; ++k) A[k] = load_a_frag(wbase, N_, i0, k * 32, lane);

    const int rowM = (lane & 16) ? 8 : 0;
    const int ncol = lane & 15;

    for (int ss = 0; ss < 16; ++ss) {
        const int s = s0 + ss;
        const bf16_t* vbase = vT + ((size_t)h * S_ + s) * C_ * N_;  // [c][j]
        if (ss + 1 < 16) {
            // warm the first cacheline of each c-row of the next s tile
            const bf16_t* nb = vbase + (size_t)C_ * N_;
            __builtin_prefetch((const void*)(nb + (size_t)lane * N_), 0, 3);
        }
        v8f acc0 = {}, acc1 = {};
#pragma unroll
        for (int k = 0; k < 12; ++k) {
            v16bf B0 = load_b_frag(vbase, N_, 0,  k * 32, lane);
            v16bf B1 = load_b_frag(vbase, N_, 16, k * 32, lane);
            acc0 = wmma_bf16(A[k], B0, acc0);
            acc1 = wmma_bf16(A[k], B1, acc1);
        }
#pragma unroll
        for (int e = 0; e < 8; ++e) {
            const int i = i0 + rowM + e;
            const size_t gb = ((size_t)s * N_ + i) * HC_ + h * 32;
            const size_t g0 = gb + ncol, g1 = gb + 16 + ncol;
            gbuf[g0] = f2bf(bf2f(gbuf[g0]) * acc0[e]);   // gated = g * o
            gbuf[g1] = f2bf(bf2f(gbuf[g1]) * acc1[e]);
        }
    }
}

// ================================================================ stage D
// out[(s,i)][cm] = gated[(s,i)][0..255] @ w_out[256][64], fp32 output.
__global__ void __launch_bounds__(256)
stage_d(const bf16_t* __restrict__ gated, const float* __restrict__ w_out,
        float* __restrict__ out)
{
    __shared__ __align__(32) bf16_t woT[CM_ * HC_];     // [n][k] 32KB
    const int tid = threadIdx.x;
    for (int idx = tid; idx < CM_ * HC_; idx += 256) {
        const int nn = idx >> 8, k = idx & 255;
        woT[idx] = f2bf(w_out[k * CM_ + nn]);
    }
    __syncthreads();

    const int wave = tid >> 5, lane = tid & 31;
    const int r0 = blockIdx.x * 128 + wave * 16;
    v16bf A[8];
#pragma unroll
    for (int k = 0; k < 8; ++k) A[k] = load_a_frag(gated, HC_, r0, k * 32, lane);

    const int rowM = (lane & 16) ? 8 : 0;
    const int ncol = lane & 15;
    for (int n = 0; n < 4; ++n) {
        v8f acc = {};
#pragma unroll
        for (int k = 0; k < 8; ++k)
            acc = wmma_bf16(A[k], load_b_frag(woT, HC_, n * 16, k * 32, lane), acc);
#pragma unroll
        for (int e = 0; e < 8; ++e)
            out[(size_t)(r0 + rowM + e) * CM_ + n * 16 + ncol] = acc[e];
    }
}

// ================================================================ launcher
extern "C" void kernel_launch(void* const* d_in, const int* in_sizes, int n_in,
                              void* d_out, int out_size, void* d_ws, size_t ws_size,
                              hipStream_t stream) {
    (void)in_sizes; (void)n_in; (void)out_size; (void)ws_size;
    const float* m        = (const float*)d_in[0];
    const float* z        = (const float*)d_in[1];
    const int*   mask     = (const int*)  d_in[2];
    const float* ln_m_w   = (const float*)d_in[3];
    const float* ln_m_b   = (const float*)d_in[4];
    const float* ln_z_w   = (const float*)d_in[5];
    const float* ln_z_b   = (const float*)d_in[6];
    const float* w_v      = (const float*)d_in[7];
    const float* w_b      = (const float*)d_in[8];
    const float* w_g      = (const float*)d_in[9];
    const float* w_out    = (const float*)d_in[10];
    float*       out      = (float*)d_out;

    // workspace layout (256B aligned chunks)
    char* ws = (char*)d_ws;
    const size_t szWb  = (size_t)H_ * N_ * N_ * sizeof(bf16_t);      //  2.36 MB
    const size_t szMln = (size_t)S_ * N_ * CM_ * sizeof(bf16_t);     // 12.6  MB
    const size_t szVT  = (size_t)H_ * S_ * C_ * N_ * sizeof(bf16_t); // 50.3  MB
    auto rnd = [](size_t x) { return (x + 255) & ~(size_t)255; };
    bf16_t* Wb   = (bf16_t*)(ws);
    bf16_t* mln  = (bf16_t*)(ws + rnd(szWb));
    bf16_t* vT   = (bf16_t*)(ws + rnd(szWb) + rnd(szMln));
    bf16_t* gbuf = (bf16_t*)(ws + rnd(szWb) + rnd(szMln) + rnd(szVT));

    stage_a <<<N_, 256, 0, stream>>>(z, mask, ln_z_w, ln_z_b, w_b, Wb);
    stage_b1<<<(S_ * N_) / 8, 256, 0, stream>>>(m, ln_m_w, ln_m_b, (uint32_t*)mln);
    stage_b2<<<(S_ * N_) / 128, 256, 0, stream>>>(mln, w_v, w_g, vT, gbuf);
    stage_c <<<dim3(S_ / 16, N_ / 16), 256, 0, stream>>>(Wb, vT, gbuf);
    stage_d <<<(S_ * N_) / 128, 256, 0, stream>>>(gbuf, w_out, out);
}
